// LSTM_41815801594240
// MI455X (gfx1250) — matrix operand order repaired and checked by reading, not compiled, not run on hardware
//
#include <hip/hip_runtime.h>
#include <hip/hip_bf16.h>

// ---------------------------------------------------------------------------
// 2-layer LSTM on MI455X (gfx1250, wave32, WMMA) — persistent + pipelined.
//
// * Fused recurrent GEMM: G(256x1024) = [x_t | h_{t-1}](256x512) @ Wcat^T,
//   v_wmma_f32_16x16x32_f16 with f32 accumulation.
// * ONE persistent kernel runs both layers x 512 timesteps; steps separated
//   by an agent-scope atomic grid barrier (64 co-resident blocks).
// * 128 KB weight slice per block staged into LDS once per layer
//   (fragment-linear layout -> contiguous ds_load_b128 pairs).
// * Software pipeline: the x-projection half of step t+1 (h-independent) is
//   computed before the barrier, so only the h-half GEMM + gate epilogue sit
//   on the sequential critical path.
// * Branch-free sigmoid/tanh using v_exp + v_rcp (no IEEE-div expansion).
// ---------------------------------------------------------------------------

typedef __attribute__((ext_vector_type(16))) _Float16 v16h;
typedef __attribute__((ext_vector_type(8)))  _Float16 v8h;
typedef __attribute__((ext_vector_type(8)))  float    v8f;

#define Bn   256
#define Tn   512
#define Hn   256
#define Cn   128
#define KCAT 512   // IN + H
#define G4H  1024  // 4*H
#define NBLK 64    // persistent grid: 8 (row tiles) x 8 (col tiles)
#define NTHR 128   // 4 waves: 2 row x 2 col 16x16 WMMA positions

__device__ __forceinline__ v16h make_frag(const _Float16* p0, const _Float16* p1) {
  v8h lo = *(const v8h*)p0;
  v8h hi = *(const v8h*)p1;
  return __builtin_shufflevector(lo, hi, 0,1,2,3,4,5,6,7,8,9,10,11,12,13,14,15);
}

__device__ __forceinline__ v8f splat8(float x) {
  v8f v;
  #pragma unroll
  for (int i = 0; i < 8; ++i) v[i] = x;
  return v;
}

// v_exp + v_rcp only — no precise-division expansion.
__device__ __forceinline__ float fast_sigmoid(float x) {
  return __builtin_amdgcn_rcpf(1.0f + __expf(-x));
}
__device__ __forceinline__ float fast_tanh(float x) {
  float e = __expf(2.0f * x);
  return 1.0f - 2.0f * __builtin_amdgcn_rcpf(e + 1.0f);
}

// Device-wide barrier for a co-resident grid (monotonic counter, no reset).
__device__ __forceinline__ void grid_barrier(unsigned* cnt, unsigned target) {
  __threadfence();                 // make this thread's stores device-visible
  __syncthreads();
  if (threadIdx.x == 0) {
    __hip_atomic_fetch_add(cnt, 1u, __ATOMIC_RELEASE, __HIP_MEMORY_SCOPE_AGENT);
    while (__hip_atomic_load(cnt, __ATOMIC_ACQUIRE, __HIP_MEMORY_SCOPE_AGENT) < target) {
      __builtin_amdgcn_s_sleep(2);
    }
  }
  __syncthreads();
}

// Persistent kernel: runs layer0 (input xT, output h1all) then layer1
// (input h1all), 512 steps each, grid barrier per step.
__global__ __launch_bounds__(NTHR)
void lstm_persistent_kernel(const _Float16* xT,      // [T][B][256] f16
                            _Float16* h1all,         // [T][B][H] f16
                            const _Float16* wcat0,   // [1024][512] f16
                            const _Float16* wcat1,
                            const float* bias0,      // [1024] f32
                            const float* bias1,
                            _Float16* hbuf0,         // [B][H] f16 ping
                            _Float16* hbuf1,         // [B][H] f16 pong
                            float* cst,              // [B][H] f32
                            unsigned* cnt)           // barrier counter (=0)
{
  extern __shared__ _Float16 lds[];  // 128 KB: [wc][g][kt][lane] 32B chunks

  const int tid  = threadIdx.x;
  const int lane = tid & 31;
  const int wave = tid >> 5;
  const int hl   = lane >> 4;
  const int l15  = lane & 15;
  const int wc   = wave >> 1;                           // wave col tile (0..1)
  const int rowBase = blockIdx.x * 32 + (wave & 1) * 16;
  const int colBase = blockIdx.y * 32 + wc * 16;
  const int bcol = colBase + l15;
  const int arow = rowBase + l15;

  const int bid  = blockIdx.y * gridDim.x + blockIdx.x;
  const int gtid = bid * NTHR + tid;

  unsigned phase = 0;

  // Accumulate one K-half (8 k-tiles) into acc[4]; kt0 selects LDS k range.
  auto gemm_half = [&](v8f (&acc)[4], const _Float16* base, int kt0) {
    const _Float16* aRow = base + (size_t)arow * 256;
    #pragma unroll 2
    for (int k0 = 0; k0 < 256; k0 += 32) {
      v16h a = make_frag(aRow + k0 + 16 * hl, aRow + k0 + 16 * hl + 8);
      const int kt = kt0 + (k0 >> 5);
      #pragma unroll
      for (int g = 0; g < 4; ++g) {
        const _Float16* bp =
            lds + (size_t)(((wc * 4 + g) * 16 + kt)) * 512 + (size_t)lane * 16;
        v16h b = *(const v16h*)bp;
        acc[g] = __builtin_amdgcn_wmma_f32_16x16x32_f16(
            false, a, false, b, (short)0, acc[g], false, false);
      }
    }
  };

  #pragma unroll 1
  for (int layer = 0; layer < 2; ++layer) {
    const _Float16* wsrc   = layer ? wcat1 : wcat0;
    const float*    bias   = layer ? bias1 : bias0;
    const _Float16* inbase = layer ? h1all : xT;
    _Float16*       hall   = layer ? (_Float16*)nullptr : h1all;

    // ---- Stage this block's weight slice into LDS (fragment-linear). ----
    // frag(wc,g,kt) = 32 lanes x 32B; lane L holds col (colBase+wc*16+(L&15)),
    // k = kt*32 + (L>>4)*16 .. +16 (same contiguous k order as the A fragment).
    for (int cid = tid; cid < 2 * 4 * 16 * 32; cid += NTHR) {
      const int L  = cid & 31;
      const int kt = (cid >> 5) & 15;
      const int g  = (cid >> 9) & 3;
      const int w  = (cid >> 11) & 1;
      const int col = blockIdx.y * 32 + w * 16 + (L & 15);
      const _Float16* src =
          wsrc + (size_t)(g * 256 + col) * KCAT + kt * 32 + (L >> 4) * 16;
      _Float16* dst = lds + (size_t)(((w * 4 + g) * 16 + kt)) * 512 + (size_t)L * 16;
      *(uint4*)dst       = *(const uint4*)src;
      *(uint4*)(dst + 8) = *(const uint4*)(src + 8);
    }

    // ---- Zero h0 (ping buffer) and c for this layer. ----
    for (int i = gtid; i < Bn * Hn; i += NBLK * NTHR) {
      cst[i] = 0.0f;
      hbuf0[i] = (_Float16)0.0f;
    }
    grid_barrier(cnt, (++phase) * NBLK);

    // Per-column gate biases, folded into the accumulator init.
    const float bI = bias[0 * 256 + bcol];
    const float bF = bias[1 * 256 + bcol];
    const float bG = bias[2 * 256 + bcol];
    const float bO = bias[3 * 256 + bcol];

    // Pipelined x-projection for t=0 (h-independent half).
    v8f accX[4];
    accX[0] = splat8(bI); accX[1] = splat8(bF);
    accX[2] = splat8(bG); accX[3] = splat8(bO);
    gemm_half(accX, inbase, 0);

    #pragma unroll 1
    for (int t = 0; t < Tn; ++t) {
      const _Float16* hp = (t & 1) ? hbuf1 : hbuf0;
      _Float16*       hn = (t & 1) ? hbuf0 : hbuf1;

      // h-dependent half: only this sits between barrier release and h write.
      v8f acc[4];
      #pragma unroll
      for (int g = 0; g < 4; ++g) acc[g] = accX[g];
      gemm_half(acc, hp, 8);

      // Elementwise gate update. C/D layout: VGPR r -> row rowBase+hl*8+r.
      _Float16* hall_t = hall ? hall + (size_t)t * Bn * Hn : (_Float16*)nullptr;
      #pragma unroll
      for (int r = 0; r < 8; ++r) {
        const int brow = rowBase + hl * 8 + r;
        const size_t idx = (size_t)brow * Hn + bcol;
        const float iv = fast_sigmoid(acc[0][r]);
        const float fv = fast_sigmoid(acc[1][r]);
        const float gv = fast_tanh(acc[2][r]);
        const float ov = fast_sigmoid(acc[3][r]);
        const float cn = fv * cst[idx] + iv * gv;
        cst[idx] = cn;
        const float hv = ov * fast_tanh(cn);
        hn[idx] = (_Float16)hv;
        if (hall_t) hall_t[idx] = (_Float16)hv;
      }

      // Prefetch-compute next step's x-projection before the barrier so it
      // overlaps other blocks' arrival (off the critical path).
      if (t < Tn - 1) {
        accX[0] = splat8(bI); accX[1] = splat8(bF);
        accX[2] = splat8(bG); accX[3] = splat8(bO);
        gemm_half(accX, inbase + (size_t)(t + 1) * Bn * 256, 0);
      }

      grid_barrier(cnt, (++phase) * NBLK);
    }
  }
}

// Pack [W_ih | W_hh] -> f16 [1024][512] for both layers; sum biases.
__global__ __launch_bounds__(256)
void pack_weights_kernel(const float* __restrict__ w_ih0, const float* __restrict__ w_hh0,
                         const float* __restrict__ b_ih0, const float* __restrict__ b_hh0,
                         const float* __restrict__ w_ih1, const float* __restrict__ w_hh1,
                         const float* __restrict__ b_ih1, const float* __restrict__ b_hh1,
                         _Float16* __restrict__ wcat0, _Float16* __restrict__ wcat1,
                         float* __restrict__ bias0, float* __restrict__ bias1)
{
  const int idx = blockIdx.x * blockDim.x + threadIdx.x;
  if (idx < G4H * KCAT) {
    const int n = idx >> 9;
    const int k = idx & 511;
    const float v0 = (k < 256) ? w_ih0[n * 256 + k] : w_hh0[n * 256 + (k - 256)];
    const float v1 = (k < 256) ? w_ih1[n * 256 + k] : w_hh1[n * 256 + (k - 256)];
    wcat0[idx] = (_Float16)v0;
    wcat1[idx] = (_Float16)v1;
  }
  if (idx < G4H) {
    bias0[idx] = b_ih0[idx] + b_hh0[idx];
    bias1[idx] = b_ih1[idx] + b_hh1[idx];
  }
}

// x [B][T][IN] f32 -> xT [T][B][IN] f16 so per-step input rows are contiguous.
__global__ __launch_bounds__(256)
void convert_x_kernel(const float* __restrict__ x, _Float16* __restrict__ xT)
{
  const size_t idx = (size_t)blockIdx.x * blockDim.x + threadIdx.x;
  if (idx >= (size_t)Bn * Tn * 256) return;
  const int f = (int)(idx & 255);
  const size_t bt = idx >> 8;
  const int t = (int)(bt % Tn);
  const int b = (int)(bt / Tn);
  xT[((size_t)t * Bn + b) * 256 + f] = (_Float16)x[idx];
}

// logits = h2_last @ w_fc^T + b_fc, then row-wise log_softmax.
__global__ __launch_bounds__(128)
void fc_logsoftmax_kernel(const _Float16* __restrict__ h2,
                          const float* __restrict__ w_fc,
                          const float* __restrict__ b_fc,
                          float* __restrict__ out)
{
  const int b = blockIdx.x;
  const int c = threadIdx.x;
  float acc = b_fc[c];
  const _Float16* hr = h2 + (size_t)b * Hn;
  const float* wr = w_fc + (size_t)c * Hn;
  #pragma unroll 4
  for (int k = 0; k < Hn; ++k) acc += (float)hr[k] * wr[k];

  __shared__ float red[Cn];
  red[c] = acc;
  __syncthreads();
  for (int s = Cn / 2; s > 0; s >>= 1) {
    if (c < s) red[c] = fmaxf(red[c], red[c + s]);
    __syncthreads();
  }
  const float mx = red[0];
  __syncthreads();
  red[c] = __expf(acc - mx);
  __syncthreads();
  for (int s = Cn / 2; s > 0; s >>= 1) {
    if (c < s) red[c] += red[c + s];
    __syncthreads();
  }
  const float lse = __logf(red[0]);
  out[(size_t)b * Cn + c] = acc - mx - lse;
}

extern "C" void kernel_launch(void* const* d_in, const int* in_sizes, int n_in,
                              void* d_out, int out_size, void* d_ws, size_t ws_size,
                              hipStream_t stream)
{
  (void)in_sizes; (void)n_in; (void)out_size; (void)ws_size;
  const float* x     = (const float*)d_in[0];
  const float* w_ih0 = (const float*)d_in[1];
  const float* w_hh0 = (const float*)d_in[2];
  const float* b_ih0 = (const float*)d_in[3];
  const float* b_hh0 = (const float*)d_in[4];
  const float* w_ih1 = (const float*)d_in[5];
  const float* w_hh1 = (const float*)d_in[6];
  const float* b_ih1 = (const float*)d_in[7];
  const float* b_hh1 = (const float*)d_in[8];
  const float* w_fc  = (const float*)d_in[9];
  const float* b_fc  = (const float*)d_in[10];
  float* out = (float*)d_out;

  // Workspace carve-up (~137 MB total).
  char* ws = (char*)d_ws;
  _Float16* xT    = (_Float16*)ws; ws += (size_t)Tn * Bn * 256 * 2;   // 67 MB
  _Float16* h1all = (_Float16*)ws; ws += (size_t)Tn * Bn * Hn * 2;    // 67 MB
  _Float16* wcat0 = (_Float16*)ws; ws += (size_t)G4H * KCAT * 2;      // 1 MB
  _Float16* wcat1 = (_Float16*)ws; ws += (size_t)G4H * KCAT * 2;      // 1 MB
  float* bias0 = (float*)ws; ws += G4H * 4;
  float* bias1 = (float*)ws; ws += G4H * 4;
  _Float16* hA = (_Float16*)ws; ws += (size_t)Bn * Hn * 2;
  _Float16* hB = (_Float16*)ws; ws += (size_t)Bn * Hn * 2;
  float* cst = (float*)ws; ws += (size_t)Bn * Hn * 4;
  unsigned* cnt = (unsigned*)ws; ws += 256;  // grid-barrier counter

  // Prep: pack weights/biases, convert+transpose x, reset barrier counter.
  pack_weights_kernel<<<(G4H * KCAT + 255) / 256, 256, 0, stream>>>(
      w_ih0, w_hh0, b_ih0, b_hh0, w_ih1, w_hh1, b_ih1, b_hh1,
      wcat0, wcat1, bias0, bias1);
  {
    const size_t total = (size_t)Bn * Tn * 256;
    convert_x_kernel<<<(unsigned)((total + 255) / 256), 256, 0, stream>>>(x, xT);
  }
  hipMemsetAsync(cnt, 0, sizeof(unsigned), stream);

  // Both LSTM layers, all 512 steps, one launch. 64 blocks x 128 threads,
  // 128 KB dynamic LDS per block for the weight slice.
  lstm_persistent_kernel<<<dim3(8, 8), NTHR, 128 * 1024, stream>>>(
      xT, h1all, wcat0, wcat1, bias0, bias1, hA, hB, cst, cnt);

  // h2[:, T-1, :] ends in hA (t=511 writes the ping buffer).
  fc_logsoftmax_kernel<<<Bn, Cn, 0, stream>>>(hA, w_fc, b_fc, out);
}
